// ODEGRU_29592324669883
// MI455X (gfx1250) — matrix operand order repaired and checked by reading, not compiled, not run on hardware
//
#include <hip/hip_runtime.h>

// ---------------------------------------------------------------------------
// ODE-GRU for MI455X (gfx1250): persistent wave32 WMMA kernel.
// B=64, L=512, I=256, H=512, HD=512, GRID_SIZE=2 (one RK4-3/8 step / t).
// 4 workgroups x 16 batch rows; 16 waves/WG, each owns 2 feature tiles (32
// hidden features). Hidden state + RK4 k's live in v8f register fragments;
// LDS stages bf16 A-operands; weights stream from L2 in pre-packed bf16
// B-fragment layout. GEMM kt-loops are ROLLED (#pragma unroll 1) with a
// wave-uniform running byte offset so B loads select the global SADDR form;
// activations use straight-line v_exp_f32/v_rcp_f32 (no branchy libm).
// ---------------------------------------------------------------------------

#define Bc  64
#define Lc  512
#define Ic  256
#define Hc  512
#define HDc 512

typedef __bf16 bf16_t;
typedef __attribute__((ext_vector_type(16))) __bf16 v16bf;
typedef __attribute__((ext_vector_type(8)))  __bf16 v8bf;
typedef __attribute__((ext_vector_type(8)))  float  v8f;

#define YS 520   // LDS row stride (halfs) for H-wide buffers: 260 dwords, %64==4
#define XS 264   // LDS row stride (halfs) for I-wide buffer

union ABu { v16bf v; v8bf h[2]; };

// Fast straight-line activations: v_exp_f32 is exp2 on AMDGPU.
// tanh(x) = 1 - 2/(e^{2x}+1); saturates to +/-1 as exp2 -> inf/0.
static __device__ __forceinline__ float fast_tanh(float x) {
    float e = __builtin_amdgcn_exp2f(x * 2.885390081777927f);   // e^{2x}
    return 1.f - 2.f * __builtin_amdgcn_rcpf(e + 1.f);
}
static __device__ __forceinline__ float fast_sigmoid(float x) {
    float e = __builtin_amdgcn_exp2f(x * -1.4426950408889634f); // e^{-x}
    return __builtin_amdgcn_rcpf(1.f + e);
}

// Multi-accumulator GEMM with ROLLED k-loop.
// A fragment (16x32 bf16, M x K) in contiguous k order (B packed the same):
//   lane l: m = l%16, hsel = l/16; elems 0..15 -> K = kt*32 + hsel*16 + e
// B fragments are fragment-linear packed: 512 halfs per (nt,kt) fragment,
// lane*16 contiguous halfs per lane (32B coalesced L2 load). B addresses are
// global-base + per-acc constant offset + wave-uniform running offset so the
// backend can use the global_load saddr+voffset form.
template <int NACC>
static __device__ __forceinline__ void gemm_multi(v8f* acc, const bf16_t* A, int astride,
                                                  const bf16_t* Bf, const int* nts, int KT,
                                                  int mrow, int hsel, int lane) {
    const bf16_t* bbase = Bf + (size_t)lane * 16;   // per-lane 32B slice
    size_t boff[NACC];
    #pragma unroll
    for (int j = 0; j < NACC; ++j)
        boff[j] = (size_t)nts[j] * KT * 512;        // fragment-linear base (uniform)
    const bf16_t* ap = A + mrow * astride + hsel * 16;

    size_t koff = 0;                                 // uniform running offset
    #pragma unroll 1
    for (int kt = 0; kt < KT; ++kt) {
        ABu u;
        u.h[0] = *(const v8bf*)(ap);
        u.h[1] = *(const v8bf*)(ap + 8);
        ap += 32;
        #pragma unroll
        for (int j = 0; j < NACC; ++j) {
            v16bf b = *(const v16bf*)(bbase + boff[j] + koff);
            acc[j] = __builtin_amdgcn_wmma_f32_16x16x32_bf16(
                         false, u.v, false, b, (short)0, acc[j], false, false);
        }
        koff += 512;
    }
}

// C/D fragment layout: lane -> col = tile*16 + lane%16; VGPR r -> row r + 8*(lane/16)
static __device__ __forceinline__ void store_frag_bf16(bf16_t* buf, int stride, int nt,
                                                       v8f f, int lane) {
    int col  = nt * 16 + (lane & 15);
    int hsel = lane >> 4;
    #pragma unroll
    for (int r = 0; r < 8; ++r)
        buf[(r + 8 * hsel) * stride + col] = (bf16_t)f[r];
}

static __device__ __forceinline__ v8f splat8(float v) {
    v8f a;
    #pragma unroll
    for (int r = 0; r < 8; ++r) a[r] = v;
    return a;
}

// ---------------------------------------------------------------------------
// Weight repack: fp32 (N x K) row-major -> bf16 fragment-linear B layout.
// g = frag*512 + lane*16 + e ; n = nt*16 + lane%16 ; k = kt*32 + (lane/16)*16 + e
// ---------------------------------------------------------------------------
__global__ void pack_weights(const float* __restrict__ W, bf16_t* __restrict__ out,
                             int Kc, int KT, size_t total) {
    for (size_t g = (size_t)blockIdx.x * blockDim.x + threadIdx.x; g < total;
         g += (size_t)gridDim.x * blockDim.x) {
        size_t f = g >> 9;
        int r  = (int)(g & 511);
        int ln = r >> 4;
        int e  = r & 15;
        int nt = (int)(f / KT);
        int kt = (int)(f % KT);
        int n  = nt * 16 + (ln & 15);
        int k  = kt * 32 + (ln >> 4) * 16 + e;
        out[g] = (bf16_t)W[(size_t)n * Kc + k];
    }
}

// ---------------------------------------------------------------------------
// Main persistent kernel: grid=4 (16 batch rows each), block=512 (16 waves).
// ---------------------------------------------------------------------------
__global__ __launch_bounds__(512, 1)
void odegru_kernel(const float* __restrict__ x, const float* __restrict__ td,
                   const float* __restrict__ b_ih, const float* __restrict__ b_hh,
                   const float* __restrict__ db0, const float* __restrict__ db1,
                   const float* __restrict__ h0, const int* __restrict__ seqlen,
                   const bf16_t* __restrict__ wihf, const bf16_t* __restrict__ whhf,
                   const bf16_t* __restrict__ dw0f, const bf16_t* __restrict__ dw1f,
                   float* __restrict__ outputs, float* __restrict__ finalo)
{
    __shared__ bf16_t ybuf[16 * YS];   // A staging: y / h / h_ode (bf16)
    __shared__ bf16_t tbuf[16 * YS];   // A staging: MLP layer-0 output
    __shared__ bf16_t xbuf[16 * XS];   // A staging: x_t (bf16, masked)
    __shared__ float  tdbuf[16];
    __shared__ int    slds[16];

    const int tid  = threadIdx.x;
    const int lane = tid & 31;
    const int wid  = tid >> 5;          // 0..15
    const int nl   = lane & 15;         // column / A-row within tile
    const int hsel = lane >> 4;
    const int b0   = blockIdx.x * 16;   // first batch row of this WG

    const int nt0 = wid * 2;            // this wave's two feature tiles
    const int nt1 = wid * 2 + 1;
    const int nts2[2] = {nt0, nt1};

    if (tid < 16) slds[tid] = seqlen[b0 + tid];

    // Hidden state fragments (persist in registers across all 512 timesteps)
    v8f hreg[2];
    hreg[0] = splat8(h0[nt0 * 16 + nl]);
    hreg[1] = splat8(h0[nt1 * 16 + nl]);
    __syncthreads();

    #pragma unroll 1
    for (int t = 0; t < Lc; ++t) {
        // ---- stage masked x_t / td_t, and bf16(h) as y0 ----
        for (int idx = tid; idx < 16 * Ic; idx += 512) {
            int m = idx >> 8;
            int c = idx & 255;
            float xv = (t < slds[m]) ? x[((size_t)(b0 + m) * Lc + t) * Ic + c] : 0.f;
            xbuf[m * XS + c] = (bf16_t)xv;
        }
        if (tid < 16)
            tdbuf[tid] = (t < slds[tid]) ? td[(size_t)(b0 + tid) * Lc + t] : 0.f;
        store_frag_bf16(ybuf, YS, nt0, hreg[0], lane);
        store_frag_bf16(ybuf, YS, nt1, hreg[1], lane);
        __syncthreads();

        float tdv[8];
        #pragma unroll
        for (int r = 0; r < 8; ++r) tdv[r] = tdbuf[r + 8 * hsel];

        // ---- one f evaluation: k = tanh(tanh(y@dw0^T+db0)@dw1^T+db1)*td ----
        auto feval = [&](v8f* kout) {
            v8f acc[2];
            acc[0] = splat8(db0[nt0 * 16 + nl]);
            acc[1] = splat8(db0[nt1 * 16 + nl]);
            gemm_multi<2>(acc, ybuf, YS, dw0f, nts2, 16, nl, hsel, lane);
            #pragma unroll
            for (int j = 0; j < 2; ++j) {
                v8f tv;
                #pragma unroll
                for (int r = 0; r < 8; ++r) tv[r] = fast_tanh(acc[j][r]);
                store_frag_bf16(tbuf, YS, nts2[j], tv, lane);
            }
            __syncthreads();
            acc[0] = splat8(db1[nt0 * 16 + nl]);
            acc[1] = splat8(db1[nt1 * 16 + nl]);
            gemm_multi<2>(acc, tbuf, YS, dw1f, nts2, 16, nl, hsel, lane);
            #pragma unroll
            for (int j = 0; j < 2; ++j)
                #pragma unroll
                for (int r = 0; r < 8; ++r) kout[j][r] = fast_tanh(acc[j][r]) * tdv[r];
        };

        v8f k1[2], k2[2], k3[2], k4[2];

        feval(k1);                                   // y1 = h
        #pragma unroll
        for (int j = 0; j < 2; ++j) {                // y2 = h + k1/3
            v8f y;
            #pragma unroll
            for (int r = 0; r < 8; ++r) y[r] = hreg[j][r] + k1[j][r] * (1.f / 3.f);
            store_frag_bf16(ybuf, YS, nts2[j], y, lane);
        }
        __syncthreads();
        feval(k2);
        #pragma unroll
        for (int j = 0; j < 2; ++j) {                // y3 = h + (k2 - k1/3)
            v8f y;
            #pragma unroll
            for (int r = 0; r < 8; ++r)
                y[r] = hreg[j][r] + k2[j][r] - k1[j][r] * (1.f / 3.f);
            store_frag_bf16(ybuf, YS, nts2[j], y, lane);
        }
        __syncthreads();
        feval(k3);
        #pragma unroll
        for (int j = 0; j < 2; ++j) {                // y4 = h + (k1 - k2 + k3)
            v8f y;
            #pragma unroll
            for (int r = 0; r < 8; ++r)
                y[r] = hreg[j][r] + k1[j][r] - k2[j][r] + k3[j][r];
            store_frag_bf16(ybuf, YS, nts2[j], y, lane);
        }
        __syncthreads();
        feval(k4);

        // h_ode = h + (k1 + 3(k2+k3) + k4)/8   (dt = 1, GRID_SIZE = 2)
        v8f hode[2];
        #pragma unroll
        for (int j = 0; j < 2; ++j) {
            #pragma unroll
            for (int r = 0; r < 8; ++r)
                hode[j][r] = hreg[j][r] +
                    (k1[j][r] + 3.f * (k2[j][r] + k3[j][r]) + k4[j][r]) * 0.125f;
            store_frag_bf16(ybuf, YS, nts2[j], hode[j], lane);
        }
        __syncthreads();

        // ---- GRU cell ----
        // Gate columns for feature tile ft: r -> ft, z -> ft+32, n -> ft+64.
        const int ntsg[6] = {nt0, nt0 + 32, nt0 + 64, nt1, nt1 + 32, nt1 + 64};

        v8f gx[6];                                   // x-part: R0,Z0,I0,R1,Z1,I1
        gx[0] = splat8(b_ih[nt0 * 16 + nl]);
        gx[1] = splat8(b_ih[Hc + nt0 * 16 + nl]);
        gx[2] = splat8(b_ih[2 * Hc + nt0 * 16 + nl]);
        gx[3] = splat8(b_ih[nt1 * 16 + nl]);
        gx[4] = splat8(b_ih[Hc + nt1 * 16 + nl]);
        gx[5] = splat8(b_ih[2 * Hc + nt1 * 16 + nl]);
        gemm_multi<6>(gx, xbuf, XS, wihf, ntsg, 8, nl, hsel, lane);

        v8f gh[6];                                   // h-part: R0,Z0,N0,R1,Z1,N1
        gh[0] = splat8(b_hh[nt0 * 16 + nl]);
        gh[1] = splat8(b_hh[Hc + nt0 * 16 + nl]);
        gh[2] = splat8(b_hh[2 * Hc + nt0 * 16 + nl]);
        gh[3] = splat8(b_hh[nt1 * 16 + nl]);
        gh[4] = splat8(b_hh[Hc + nt1 * 16 + nl]);
        gh[5] = splat8(b_hh[2 * Hc + nt1 * 16 + nl]);
        gemm_multi<6>(gh, ybuf, YS, whhf, ntsg, 16, nl, hsel, lane);

        #pragma unroll
        for (int j = 0; j < 2; ++j) {
            int col = nts2[j] * 16 + nl;
            #pragma unroll
            for (int r = 0; r < 8; ++r) {
                float rr = fast_sigmoid(gx[3 * j + 0][r] + gh[3 * j + 0][r]);
                float zz = fast_sigmoid(gx[3 * j + 1][r] + gh[3 * j + 1][r]);
                float nn = fast_tanh(gx[3 * j + 2][r] + rr * gh[3 * j + 2][r]);
                float hn = (1.f - zz) * nn + zz * hode[j][r];
                hreg[j][r] = hn;
                int brow = b0 + r + 8 * hsel;
                outputs[((size_t)brow * Lc + t) * Hc + col] = hn;
                if (t == slds[r + 8 * hsel] - 1)
                    finalo[(size_t)brow * Hc + col] = hn;
            }
        }
        __syncthreads();   // before next iteration overwrites xbuf/ybuf
    }
}

// ---------------------------------------------------------------------------
extern "C" void kernel_launch(void* const* d_in, const int* in_sizes, int n_in,
                              void* d_out, int out_size, void* d_ws, size_t ws_size,
                              hipStream_t stream) {
    const float* x    = (const float*)d_in[0];
    const float* td   = (const float*)d_in[1];
    const float* w_ih = (const float*)d_in[2];
    const float* w_hh = (const float*)d_in[3];
    const float* b_ih = (const float*)d_in[4];
    const float* b_hh = (const float*)d_in[5];
    const float* dw0  = (const float*)d_in[6];
    const float* db0  = (const float*)d_in[7];
    const float* dw1  = (const float*)d_in[8];
    const float* db1  = (const float*)d_in[9];
    const float* h0   = (const float*)d_in[10];
    const int*   seq  = (const int*)d_in[11];

    // bf16 fragment-packed weights in workspace (~3.25 MB, L2-resident)
    bf16_t* ws = (bf16_t*)d_ws;
    const size_t n_wih = (size_t)96 * 8  * 512;   // 1536x256
    const size_t n_whh = (size_t)96 * 16 * 512;   // 1536x512
    const size_t n_d   = (size_t)32 * 16 * 512;   // 512x512
    bf16_t* wihf = ws;
    bf16_t* whhf = wihf + n_wih;
    bf16_t* dw0f = whhf + n_whh;
    bf16_t* dw1f = dw0f + n_d;

    pack_weights<<<512,  256, 0, stream>>>(w_ih, wihf, 256, 8,  n_wih);
    pack_weights<<<1024, 256, 0, stream>>>(w_hh, whhf, 512, 16, n_whh);
    pack_weights<<<512,  256, 0, stream>>>(dw0,  dw0f, 512, 16, n_d);
    pack_weights<<<512,  256, 0, stream>>>(dw1,  dw1f, 512, 16, n_d);

    float* outputs = (float*)d_out;                       // (B, L, H)
    float* finalo  = outputs + (size_t)Bc * Lc * Hc;      // (1, B, H)

    odegru_kernel<<<4, 512, 0, stream>>>(x, td, b_ih, b_hh, db0, db1, h0, seq,
                                         wihf, whhf, dw0f, dw1f, outputs, finalo);
}
